// LSTM_82154134438018
// MI455X (gfx1250) — matrix operand order repaired and checked by reading, not compiled, not run on hardware
//
#include <hip/hip_runtime.h>

typedef __attribute__((ext_vector_type(16))) _Float16 v16h;
typedef __attribute__((ext_vector_type(8)))  _Float16 v8h;
typedef __attribute__((ext_vector_type(8)))  float    v8f;

#define B_SZ 4096
#define E_SZ 1024
#define H_SZ 2048
#define K_SZ 3072   // E + H
#define BLK_M 64
#define BLK_N 128   // per gate
#define BLK_K 32

__device__ __forceinline__ float fast_sigmoid(float x) {
    x = fminf(fmaxf(x, -30.f), 30.f);
    return __builtin_amdgcn_rcpf(1.f + __expf(-x));
}
__device__ __forceinline__ float fast_tanh(float x) {
    x = fminf(fmaxf(x, -15.f), 15.f);
    float e = __expf(2.f * x);
    return (e - 1.f) * __builtin_amdgcn_rcpf(e + 1.f);
}

// CDNA5 async global->LDS copy (ASYNCcnt-tracked, bypasses VGPR staging).
__device__ __forceinline__ void async_copy_b128(void* lds_ptr, const void* gptr) {
    unsigned lds_addr = (unsigned)(unsigned long long)lds_ptr;
    asm volatile("global_load_async_to_lds_b128 %0, %1, off"
                 :: "v"(lds_addr), "v"(gptr)
                 : "memory");
}
__device__ __forceinline__ void wait_async0() {
    asm volatile("s_wait_asynccnt 0" ::: "memory");
}

// ---------------------------------------------------------------------------
// Pack A = [emb[x] | h]  (4096 x 3072) fp32 -> fp16, row-major.
// ---------------------------------------------------------------------------
__global__ __launch_bounds__(256)
void pack_a(const int* __restrict__ x, const float* __restrict__ h,
            const float* __restrict__ emb, _Float16* __restrict__ Ah) {
    const int t = blockIdx.x * 256 + threadIdx.x;       // 8 halves per thread
    const int row = t / (K_SZ / 8);
    if (row >= B_SZ) return;
    const int kc = (t % (K_SZ / 8)) * 8;
    const float* src = (kc < E_SZ)
        ? emb + (long)x[row] * E_SZ + kc
        : h + (long)row * H_SZ + (kc - E_SZ);
    float4 f0 = ((const float4*)src)[0];
    float4 f1 = ((const float4*)src)[1];
    v8h o;
    o[0] = (_Float16)f0.x; o[1] = (_Float16)f0.y;
    o[2] = (_Float16)f0.z; o[3] = (_Float16)f0.w;
    o[4] = (_Float16)f1.x; o[5] = (_Float16)f1.y;
    o[6] = (_Float16)f1.z; o[7] = (_Float16)f1.w;
    *(v8h*)(Ah + (long)row * K_SZ + kc) = o;
}

// ---------------------------------------------------------------------------
// Pack Wf/Wi/Wc/Wo (each 3072 x 2048, row-major) fp32 -> fp16, concatenated.
// ---------------------------------------------------------------------------
__global__ __launch_bounds__(256)
void pack_w(const float* __restrict__ Wf, const float* __restrict__ Wi,
            const float* __restrict__ Wc, const float* __restrict__ Wo,
            _Float16* __restrict__ Wh) {
    const long per = (long)K_SZ * H_SZ;
    const long t = (long)blockIdx.x * 256 + threadIdx.x;
    if (t >= 4 * per / 8) return;
    const long e = t * 8;
    const int g = (int)(e / per);
    const long off = e - (long)g * per;
    const float* W = (g == 0) ? Wf : (g == 1) ? Wi : (g == 2) ? Wc : Wo;
    float4 f0 = ((const float4*)(W + off))[0];
    float4 f1 = ((const float4*)(W + off))[1];
    v8h o;
    o[0] = (_Float16)f0.x; o[1] = (_Float16)f0.y;
    o[2] = (_Float16)f0.z; o[3] = (_Float16)f0.w;
    o[4] = (_Float16)f1.x; o[5] = (_Float16)f1.y;
    o[6] = (_Float16)f1.z; o[7] = (_Float16)f1.w;
    *(v8h*)(Wh + e) = o;
}

// ---------------------------------------------------------------------------
// Helpers for the GEMM main loop.
// ---------------------------------------------------------------------------
union U { v16h v; v8h h[2]; };

__device__ __forceinline__ void issue_tiles(_Float16* sAb, _Float16* sWb,
                                            const _Float16* gA, const _Float16* gW,
                                            int lAoff, int lWoff) {
    async_copy_b128(sAb + lAoff, gA);
#pragma unroll
    for (int q = 0; q < 4; ++q)
        async_copy_b128(sWb + lWoff + q * 8, gW + q * 8);
}

// One BLK_K step: all A/B fragments loaded into distinct registers first so the
// 16-wmma chain is not gated on per-pair LDS latency.
__device__ __forceinline__ void compute_step(const _Float16* sAb, const _Float16* sWb,
                                             v8f (&acc)[4][2][2],
                                             int waveM, int waveN, int lane) {
    const int fr = lane & 15;
    const int hi = lane >> 4;

    U a[2];
#pragma unroll
    for (int mt = 0; mt < 2; ++mt) {
        const _Float16* pa = sAb + (waveM * 32 + mt * 16 + fr) * BLK_K + hi * 8;
        a[mt].h[0] = *(const v8h*)pa;
        a[mt].h[1] = *(const v8h*)(pa + 16);
    }
    U b[4][2];
#pragma unroll
    for (int g = 0; g < 4; ++g)
#pragma unroll
        for (int nt = 0; nt < 2; ++nt) {
            const _Float16* pb = sWb + g * (BLK_K * BLK_N)
                                     + fr + waveN * 32 + nt * 16;
#pragma unroll
            for (int s = 0; s < 8; ++s) {
                b[g][nt].v[s]     = pb[(hi * 8 + s) * BLK_N];
                b[g][nt].v[8 + s] = pb[(16 + hi * 8 + s) * BLK_N];
            }
        }
#pragma unroll
    for (int g = 0; g < 4; ++g)
#pragma unroll
        for (int nt = 0; nt < 2; ++nt)
#pragma unroll
            for (int mt = 0; mt < 2; ++mt)
                acc[g][mt][nt] = __builtin_amdgcn_wmma_f32_16x16x32_f16(
                    false, a[mt].v, false, b[g][nt].v, (short)0,
                    acc[g][mt][nt], false, false);
}

// ---------------------------------------------------------------------------
// Fused 4-gate WMMA GEMM + LSTM epilogue.
// Block tile M=64 x N=128 per gate; 8 waves (2 M x 4 N); 16 wmma / K-step.
// Ping-pong LDS buffers filled by GLOBAL_LOAD_ASYNC_TO_LDS_B128; K loop
// manually unrolled x2 so buffer selection is compile-time static.
// ---------------------------------------------------------------------------
__global__ __launch_bounds__(256)
void lstm_gemm(const _Float16* __restrict__ Ah, const _Float16* __restrict__ Wh,
               const float* __restrict__ bF, const float* __restrict__ bI,
               const float* __restrict__ bC, const float* __restrict__ bO,
               const float* __restrict__ cin, float* __restrict__ out) {
    __shared__ __align__(32) _Float16 sA[2][BLK_M * BLK_K];     // 2 x 4 KB
    __shared__ __align__(32) _Float16 sW[2][4][BLK_K * BLK_N];  // 2 x 32 KB

    const int tid   = threadIdx.x;
    const int lane  = tid & 31;
    const int wave  = tid >> 5;
    const int waveM = wave & 1;
    const int waveN = wave >> 1;
    const int bm = blockIdx.y * BLK_M;
    const int bn = blockIdx.x * BLK_N;

    // cooperative-load assignments (per thread: 1 A chunk + 4 W chunks, 16B each)
    const int aRow = tid >> 2;
    const int aKc  = (tid & 3) * 8;
    const int wg   = tid >> 6;
    const int wr   = tid & 63;
    const int wK   = wr >> 1;
    const int wNc  = (wr & 1) * 64;

    const _Float16* gA = Ah + (long)(bm + aRow) * K_SZ + aKc;
    const _Float16* gW = Wh + (long)wg * ((long)K_SZ * H_SZ)
                            + (long)wK * H_SZ + bn + wNc;
    const int lAoff = aRow * BLK_K + aKc;
    const int lWoff = wg * (BLK_K * BLK_N) + wK * BLK_N + wNc;
    const long gWstep = (long)BLK_K * H_SZ;

    v8f acc[4][2][2] = {};

    // ---- prologue: fill buffer 0 ----
    issue_tiles(sA[0], &sW[0][0][0], gA, gW, lAoff, lWoff);
    gA += BLK_K; gW += gWstep;

    // K = 3072 = 48 double-steps of 2*BLK_K.
    constexpr int NIT = K_SZ / (2 * BLK_K);
#pragma unroll 1
    for (int it = 0; it < NIT; ++it) {
        // step 0: compute buf0, prefetch buf1
        wait_async0();       // my buf0 tiles landed
        __syncthreads();     // everyone's landed; everyone done reading buf1
        issue_tiles(sA[1], &sW[1][0][0], gA, gW, lAoff, lWoff);
        gA += BLK_K; gW += gWstep;
        compute_step(sA[0], &sW[0][0][0], acc, waveM, waveN, lane);

        // step 1: compute buf1, prefetch buf0
        wait_async0();
        __syncthreads();
        if (it + 1 < NIT) {
            issue_tiles(sA[0], &sW[0][0][0], gA, gW, lAoff, lWoff);
            gA += BLK_K; gW += gWstep;
        }
        compute_step(sA[1], &sW[1][0][0], acc, waveM, waveN, lane);
    }

    // ---- fused LSTM epilogue ----
    // C layout: VGPR r -> M = r + 8*hi, N = lane&15.
    const int fr = lane & 15;
    const int hi = lane >> 4;
#pragma unroll
    for (int nt = 0; nt < 2; ++nt) {
        const int n = bn + waveN * 32 + nt * 16 + fr;
        const float vbF = bF[n], vbI = bI[n], vbC = bC[n], vbO = bO[n];
#pragma unroll
        for (int mt = 0; mt < 2; ++mt) {
#pragma unroll
            for (int r = 0; r < 8; ++r) {
                const int m = bm + waveM * 32 + mt * 16 + hi * 8 + r;
                const long idx = (long)m * H_SZ + n;
                const float fg = fast_sigmoid(acc[0][mt][nt][r] + vbF);
                const float ig = fast_sigmoid(acc[1][mt][nt][r] + vbI);
                const float gg = fast_tanh   (acc[2][mt][nt][r] + vbC);
                const float og = fast_sigmoid(acc[3][mt][nt][r] + vbO);
                const float cn = fg * cin[idx] + ig * gg;
                const float hn = og * fast_tanh(cn);
                out[idx] = hn;                           // ht
                out[(long)B_SZ * H_SZ + idx] = cn;       // ct
            }
        }
    }
}

extern "C" void kernel_launch(void* const* d_in, const int* in_sizes, int n_in,
                              void* d_out, int out_size, void* d_ws, size_t ws_size,
                              hipStream_t stream) {
    const int*   x   = (const int*)  d_in[0];
    const float* h   = (const float*)d_in[1];
    const float* c   = (const float*)d_in[2];
    const float* emb = (const float*)d_in[3];
    const float* Wf  = (const float*)d_in[4];
    const float* bf  = (const float*)d_in[5];
    const float* Wi  = (const float*)d_in[6];
    const float* bi  = (const float*)d_in[7];
    const float* Wc  = (const float*)d_in[8];
    const float* bc  = (const float*)d_in[9];
    const float* Wo  = (const float*)d_in[10];
    const float* bo  = (const float*)d_in[11];

    // workspace: A_f16 (25.2 MB) then W_f16 (50.3 MB); both L2-resident.
    _Float16* Ah = (_Float16*)d_ws;
    _Float16* Wh = Ah + (size_t)B_SZ * K_SZ;

    {
        const long n8 = (long)B_SZ * K_SZ / 8;
        pack_a<<<dim3((unsigned)((n8 + 255) / 256)), 256, 0, stream>>>(x, h, emb, Ah);
    }
    {
        const long n8 = 4L * K_SZ * H_SZ / 8;
        pack_w<<<dim3((unsigned)((n8 + 255) / 256)), 256, 0, stream>>>(Wf, Wi, Wc, Wo, Wh);
    }
    {
        dim3 grid(H_SZ / BLK_N, B_SZ / BLK_M);   // (16, 64)
        lstm_gemm<<<grid, 256, 0, stream>>>(Ah, Wh, bf, bi, bc, bo, c, (float*)d_out);
    }
}
